// SampleKernel_63436666962330
// MI455X (gfx1250) — 8-bit matrix operand order repaired and checked by reading, compiled, not run on hardware
//
#include <hip/hip_runtime.h>
#include <stdint.h>


typedef __attribute__((ext_vector_type(8))) int v8i;

#define NXS 2048
#define NYS 2048
#define DIMS 28
#define KB 320
#define KSTEPS 5

__global__ void pack_onehot_kernel(const int* __restrict__ src,
                                   uint8_t* __restrict__ dst,
                                   int nsamp) {
  int t = blockIdx.x * blockDim.x + threadIdx.x;
  if (t >= nsamp * DIMS) return;
  int s = t / DIMS;
  int d = t - s * DIMS;
  int v = src[(size_t)s * DIMS + d] & 7;
  int c = (v < 3) ? 1 : ((v == 7) ? 2 : 0);
  uint8_t* row = dst + (size_t)s * KB;
  *(unsigned long long*)(row + d * 8) = 1ull << (8 * v);
  uint8_t* cp = row + DIMS * 8 + d * 3;
  cp[0] = (uint8_t)(c == 0);
  cp[1] = (uint8_t)(c == 1);
  cp[2] = (uint8_t)(c == 2);
  if (d == 0) {
    *(unsigned int*)(row + 308) = 0u;
    *(unsigned int*)(row + 312) = 0u;
    *(unsigned int*)(row + 316) = 0u;
  }
}

__global__ void __launch_bounds__(256)
dist_gemm_kernel(const uint8_t* __restrict__ Ap,
                 const uint8_t* __restrict__ Bp,
                 const float* __restrict__ sigma_p,
                 float* __restrict__ out) {
  const int lane = threadIdx.x & 31;
  const int wave = threadIdx.x >> 5;
  const int lm   = lane & 15;
  const int hi   = lane >> 4;
  const int m0 = blockIdx.y * 64  + (wave & 1) * 32;
  const int n0 = blockIdx.x * 128 + (wave >> 1) * 32;

  v8i acc[2][2] = {};

  const uint8_t* arow0 = Ap + (size_t)(m0 + lm) * KB + hi * 8;
  const uint8_t* arow1 = arow0 + (size_t)16 * KB;
  const uint8_t* brow0 = Bp + (size_t)(n0 + lm) * KB + hi * 8;
  const uint8_t* brow1 = brow0 + (size_t)16 * KB;

  for (int kt = 0; kt < KSTEPS; ++kt) {
    const int k0 = kt * 64;
    v8i a[2], b[2];
#pragma unroll
    for (int mi = 0; mi < 2; ++mi) {
      const uint8_t* r = (mi == 0 ? arow0 : arow1) + k0;
#pragma unroll
      for (int cch = 0; cch < 4; ++cch) {
        uint2 x = *(const uint2*)(r + cch * 16);
        a[mi][2 * cch]     = (int)x.x;
        a[mi][2 * cch + 1] = (int)x.y;
      }
    }
#pragma unroll
    for (int ni = 0; ni < 2; ++ni) {
      const uint8_t* r = (ni == 0 ? brow0 : brow1) + k0;
#pragma unroll
      for (int cch = 0; cch < 4; ++cch) {
        uint2 y = *(const uint2*)(r + cch * 16);
        b[ni][2 * cch]     = (int)y.x;
        b[ni][2 * cch + 1] = (int)y.y;
      }
    }
#pragma unroll
    for (int mi = 0; mi < 2; ++mi)
#pragma unroll
      for (int ni = 0; ni < 2; ++ni)
        acc[mi][ni] = __builtin_amdgcn_wmma_i32_16x16x64_iu8(
             false, a[mi],  false, b[ni], acc[mi][ni],
             false,  false);
  }

  const float sg = sigma_p[0];
  const float inv2s2 = 1.0f / (2.0f * sg * sg);
#pragma unroll
  for (int mi = 0; mi < 2; ++mi) {
#pragma unroll
    for (int ni = 0; ni < 2; ++ni) {
#pragma unroll
      for (int vv = 0; vv < 8; ++vv) {
        int row = m0 + mi * 16 + vv + 8 * hi;
        int col = n0 + ni * 16 + lm;
        float dist = (56.0f - (float)acc[mi][ni][vv]) * (1.0f / 28.0f);
        out[(size_t)row * NYS + col] = expf(-dist * dist * inv2s2);
      }
    }
  }
}

extern "C" void kernel_launch(void* const* d_in, const int* in_sizes, int n_in,
                              void* d_out, int out_size, void* d_ws, size_t ws_size,
                              hipStream_t stream) {
  (void)in_sizes; (void)n_in; (void)out_size; (void)ws_size;
  const int*   set_r = (const int*)d_in[0];
  const int*   set_t = (const int*)d_in[1];
  const float* sigma = (const float*)d_in[2];
  float*       out   = (float*)d_out;

  uint8_t* Ap = (uint8_t*)d_ws;
  uint8_t* Bp = Ap + (size_t)NXS * KB;

  pack_onehot_kernel<<<(NXS * DIMS + 255) / 256, 256, 0, stream>>>(set_r, Ap, NXS);
  pack_onehot_kernel<<<(NYS * DIMS + 255) / 256, 256, 0, stream>>>(set_t, Bp, NYS);

  dist_gemm_kernel<<<dim3(NYS / 128, NXS / 64), 256, 0, stream>>>(Ap, Bp, sigma, out);
}
